// Net_80762565034161
// MI455X (gfx1250) — 8-bit matrix operand order repaired and checked by reading, compiled, not run on hardware
//
#include <hip/hip_runtime.h>
#include <stdint.h>


typedef int v8i __attribute__((ext_vector_type(8)));

#define HDIM   384
#define IN_F   81
#define LSTRIDE 392
#define ROWS_PER_BLOCK 64
#define EPS 1e-5f

#define S1_OFF 0
#define S2_OFF (384*128)
#define S3_OFF (384*128 + 384*384)
#define E_BASE (384*128 + 2*384*384)
#define E1_OFF (E_BASE)
#define E2_OFF (E_BASE + 384*8)
#define E3_OFF (E_BASE + 2*384*8)

__device__ __forceinline__ int8_t sign8(float v) {
    return (int8_t)((v > 0.f) - (v < 0.f));
}

__global__ __launch_bounds__(256) void prep_kernel(
    const float* __restrict__ w1, const float* __restrict__ w2,
    const float* __restrict__ w3, const float* __restrict__ w4,
    const float* __restrict__ b1, const float* __restrict__ b2,
    const float* __restrict__ b3,
    const float* __restrict__ g1, const float* __restrict__ be1,
    const float* __restrict__ m1, const float* __restrict__ v1,
    const float* __restrict__ g2, const float* __restrict__ be2,
    const float* __restrict__ m2, const float* __restrict__ v2,
    const float* __restrict__ g3, const float* __restrict__ be3,
    const float* __restrict__ m3, const float* __restrict__ v3,
    uint8_t* __restrict__ ws)
{
    int tid = blockIdx.x * blockDim.x + threadIdx.x;
    int stride = gridDim.x * blockDim.x;
    int8_t* s1 = (int8_t*)(ws + S1_OFF);
    int8_t* s2 = (int8_t*)(ws + S2_OFF);
    int8_t* s3 = (int8_t*)(ws + S3_OFF);
    for (int i = tid; i < HDIM * 128; i += stride) {
        int r = i >> 7, c = i & 127;
        float v = (c < IN_F) ? w1[r * IN_F + c] : 0.f;
        s1[i] = sign8(v);
    }
    for (int i = tid; i < HDIM * HDIM; i += stride) s2[i] = sign8(w2[i]);
    for (int i = tid; i < HDIM * HDIM; i += stride) s3[i] = sign8(w3[i]);

    float2* e1 = (float2*)(ws + E1_OFF);
    float2* e2 = (float2*)(ws + E2_OFF);
    float4* e3 = (float4*)(ws + E3_OFF);
    for (int n = tid; n < HDIM; n += stride) {
        float sc1 = g1[n] * __frsqrt_rn(v1[n] + EPS);
        e1[n] = make_float2(sc1, be1[n] + (b1[n] - m1[n]) * sc1);
        float sc2 = g2[n] * __frsqrt_rn(v2[n] + EPS);
        e2[n] = make_float2(sc2, be2[n] + (b2[n] - m2[n]) * sc2);
        float sc3 = g3[n] * __frsqrt_rn(v3[n] + EPS);
        e3[n] = make_float4(sc3, be3[n] + (b3[n] - m3[n]) * sc3,
                            w4[n], w4[HDIM + n]);
    }
}

__device__ __forceinline__ v8i load_a_frag(const uint8_t* row, int kbase, int half) {
    uint64_t d0 = *(const uint64_t*)(row + kbase +  0 + half * 16);
    uint64_t d1 = *(const uint64_t*)(row + kbase +  8 + half * 16);
    uint64_t d2 = *(const uint64_t*)(row + kbase + 32 + half * 16);
    uint64_t d3 = *(const uint64_t*)(row + kbase + 40 + half * 16);
    v8i a;
    a[0] = (int)(uint32_t)d0; a[1] = (int)(uint32_t)(d0 >> 32);
    a[2] = (int)(uint32_t)d1; a[3] = (int)(uint32_t)(d1 >> 32);
    a[4] = (int)(uint32_t)d2; a[5] = (int)(uint32_t)(d2 >> 32);
    a[6] = (int)(uint32_t)d3; a[7] = (int)(uint32_t)(d3 >> 32);
    return a;
}

template <int KCNT, bool LAST>
__device__ __forceinline__ void layer(
    int wave, int half, int l15,
    const uint8_t* inbuf, uint8_t* outbuf,
    const int8_t* __restrict__ wmat, int wstride,
    const float2* __restrict__ ep2,
    const float4* __restrict__ ep4,
    float (*lg)[2][8])
{
    const uint8_t* arow0 = inbuf + (wave * 32 + l15) * LSTRIDE;
    const uint8_t* arow1 = arow0 + 16 * LSTRIDE;

    for (int grp = 0; grp < 6; ++grp) {
        v8i acc[4][2];
        v8i zero = {0, 0, 0, 0, 0, 0, 0, 0};
        #pragma unroll
        for (int t = 0; t < 4; ++t) { acc[t][0] = zero; acc[t][1] = zero; }

        #pragma unroll
        for (int kb = 0; kb < KCNT; ++kb) {
            const int kbase = kb * 64;
            v8i a0 = load_a_frag(arow0, kbase, half);
            v8i a1 = load_a_frag(arow1, kbase, half);
            #pragma unroll
            for (int t = 0; t < 4; ++t) {
                const int n = (grp * 4 + t) * 16 + l15;
                const int8_t* wr = wmat + (size_t)n * wstride + kbase + half * 16;
                uint4 q0 = *(const uint4*)(wr);
                uint4 q1 = *(const uint4*)(wr + 32);
                v8i b;
                b[0] = (int)q0.x; b[1] = (int)q0.y; b[2] = (int)q0.z; b[3] = (int)q0.w;
                b[4] = (int)q1.x; b[5] = (int)q1.y; b[6] = (int)q1.z; b[7] = (int)q1.w;
                acc[t][0] = __builtin_amdgcn_wmma_i32_16x16x64_iu8(
                    true, a0, true, b, acc[t][0], false, false);
                acc[t][1] = __builtin_amdgcn_wmma_i32_16x16x64_iu8(
                    true, a1, true, b, acc[t][1], false, false);
            }
        }

        #pragma unroll
        for (int t = 0; t < 4; ++t) {
            const int n = (grp * 4 + t) * 16 + l15;
            float sc, sh, w40 = 0.f, w41 = 0.f;
            if (LAST) {
                float4 e = ep4[n];
                sc = e.x; sh = e.y; w40 = e.z; w41 = e.w;
            } else {
                float2 e = ep2[n];
                sc = e.x; sh = e.y;
            }
            #pragma unroll
            for (int mt = 0; mt < 2; ++mt) {
                #pragma unroll
                for (int r = 0; r < 8; ++r) {
                    float h = (float)acc[t][mt][r] * sc + sh;
                    h = fminf(1.f, fmaxf(-1.f, h));
                    if (!LAST) {
                        const int m = mt * 16 + r + 8 * half;
                        outbuf[(wave * 32 + m) * LSTRIDE + n] = (uint8_t)sign8(h);
                    } else {
                        lg[mt][0][r] += h * w40;
                        lg[mt][1][r] += h * w41;
                    }
                }
            }
        }
    }
}

__global__ __launch_bounds__(64) void bnn_forward_kernel(
    const float* __restrict__ x, const uint8_t* __restrict__ ws,
    const float* __restrict__ b4, float* __restrict__ out)
{
    __shared__ __align__(16) uint8_t bufA[ROWS_PER_BLOCK * LSTRIDE];
    __shared__ __align__(16) uint8_t bufB[ROWS_PER_BLOCK * LSTRIDE];

    const int tid  = threadIdx.x;
    const int wave = tid >> 5;
    const int lane = tid & 31;
    const int half = lane >> 4;
    const int l15  = lane & 15;
    const int wrow = blockIdx.x * ROWS_PER_BLOCK + wave * 32;

    for (int r = 0; r < 32; ++r) {
        const float* xr = x + (size_t)(wrow + r) * IN_F;
        uint8_t* dst = bufA + (wave * 32 + r) * LSTRIDE;
        for (int c = lane; c < 128; c += 32) {
            float v = (c < IN_F) ? xr[c] : 0.f;
            dst[c] = (uint8_t)sign8(v);
        }
    }
    __syncthreads();

    float lg[2][2][8];
    #pragma unroll
    for (int mt = 0; mt < 2; ++mt)
        #pragma unroll
        for (int j = 0; j < 2; ++j)
            #pragma unroll
            for (int r = 0; r < 8; ++r) lg[mt][j][r] = 0.f;

    layer<2, false>(wave, half, l15, bufA, bufB, (const int8_t*)(ws + S1_OFF), 128,
                    (const float2*)(ws + E1_OFF), nullptr, lg);
    __syncthreads();
    layer<6, false>(wave, half, l15, bufB, bufA, (const int8_t*)(ws + S2_OFF), HDIM,
                    (const float2*)(ws + E2_OFF), nullptr, lg);
    __syncthreads();
    layer<6, true >(wave, half, l15, bufA, bufB, (const int8_t*)(ws + S3_OFF), HDIM,
                    nullptr, (const float4*)(ws + E3_OFF), lg);

    const float bb0 = b4[0], bb1 = b4[1];
    #pragma unroll
    for (int mt = 0; mt < 2; ++mt) {
        #pragma unroll
        for (int r = 0; r < 8; ++r) {
            float s0 = lg[mt][0][r];
            float s1 = lg[mt][1][r];
            #pragma unroll
            for (int mask = 1; mask < 16; mask <<= 1) {
                s0 += __shfl_xor(s0, mask, 32);
                s1 += __shfl_xor(s1, mask, 32);
            }
            float l0 = s0 + bb0, l1 = s1 + bb1;
            float mx = fmaxf(l0, l1);
            float lse = mx + __logf(__expf(l0 - mx) + __expf(l1 - mx));
            if (l15 == 0) {
                const int row = wrow + mt * 16 + r + 8 * half;
                float2 o; o.x = l0 - lse; o.y = l1 - lse;
                *(float2*)(out + (size_t)row * 2) = o;
            }
        }
    }
}

extern "C" void kernel_launch(void* const* d_in, const int* in_sizes, int n_in,
                              void* d_out, int out_size, void* d_ws, size_t ws_size,
                              hipStream_t stream)
{
    const float* x   = (const float*)d_in[0];
    const float* w1  = (const float*)d_in[1];
    const float* b1  = (const float*)d_in[2];
    const float* w2  = (const float*)d_in[3];
    const float* b2  = (const float*)d_in[4];
    const float* w3  = (const float*)d_in[5];
    const float* b3  = (const float*)d_in[6];
    const float* w4  = (const float*)d_in[7];
    const float* b4  = (const float*)d_in[8];
    const float* g1  = (const float*)d_in[9];
    const float* be1 = (const float*)d_in[10];
    const float* m1  = (const float*)d_in[11];
    const float* v1  = (const float*)d_in[12];
    const float* g2  = (const float*)d_in[13];
    const float* be2 = (const float*)d_in[14];
    const float* m2  = (const float*)d_in[15];
    const float* v2  = (const float*)d_in[16];
    const float* g3  = (const float*)d_in[17];
    const float* be3 = (const float*)d_in[18];
    const float* m3  = (const float*)d_in[19];
    const float* v3  = (const float*)d_in[20];

    uint8_t* ws = (uint8_t*)d_ws;
    prep_kernel<<<256, 256, 0, stream>>>(
        w1, w2, w3, w4, b1, b2, b3,
        g1, be1, m1, v1, g2, be2, m2, v2, g3, be3, m3, v3, ws);

    const int nrows   = in_sizes[0] / IN_F;
    const int nblocks = nrows / ROWS_PER_BLOCK;
    bnn_forward_kernel<<<nblocks, 64, 0, stream>>>(
        x, ws, b4, (float*)d_out);
}
